// MultiHeadedAttention_50989851738454
// MI455X (gfx1250) — matrix operand order repaired and checked by reading, not compiled, not run on hardware
//
#include <hip/hip_runtime.h>
#include <hip/hip_bf16.h>

#define HEADS   16
#define DMODEL  1024
#define DK      64
#define BATCH   4
#define SEQ     2048
#define MROWS   (BATCH * SEQ)      /* 8192 */

typedef __attribute__((ext_vector_type(16))) __bf16 v16bf;
typedef __attribute__((ext_vector_type(8)))  __bf16 v8bf;
typedef __attribute__((ext_vector_type(8)))  float  v8f;

union Frag { v16bf v; v8bf h[2]; };

#if defined(__gfx1250__) && defined(__has_builtin)
#if __has_builtin(__builtin_amdgcn_tensor_load_to_lds) && __has_builtin(__builtin_amdgcn_s_wait_tensorcnt)
#define USE_TDM 1
#endif
#endif

static __device__ __forceinline__ unsigned short f2bfu(float f) {
  unsigned u = __float_as_uint(f);
  u += 0x7FFFu + ((u >> 16) & 1u);          // round-to-nearest-even
  return (unsigned short)(u >> 16);
}
static __device__ __forceinline__ __bf16 f2bf(float f) {
  unsigned short s = f2bfu(f);
  __bf16 r;
  __builtin_memcpy(&r, &s, sizeof(r));
  return r;
}
static __device__ __forceinline__ v8f v8f_zero() {
  v8f z = {0.f,0.f,0.f,0.f,0.f,0.f,0.f,0.f};
  return z;
}

#ifdef USE_TDM
typedef __attribute__((ext_vector_type(4))) unsigned int u32x4;
typedef __attribute__((ext_vector_type(8))) int          i32x8;
typedef __attribute__((ext_vector_type(4))) int          i32x4;

// TDM load of one 128-row x 32-elem bf16 tile (row = 64B) into LDS with
// 16B padding per row (LDS stride 80B == [128][40] bf16), dim0 stride = 1024.
static __device__ __forceinline__ void tdm_load_tile(const void* gptr,
                                                     unsigned lds_addr) {
  unsigned long long ga = (unsigned long long)gptr;
  u32x4 g0;
  g0[0] = 1u;                                                   // count=1 (valid D#)
  g0[1] = lds_addr;                                             // LDS byte address
  g0[2] = (unsigned)ga;                                         // global_addr[31:0]
  g0[3] = (unsigned)((ga >> 32) & 0x01FFFFFFull) | 0x80000000u; // addr[56:32] | type=2
  i32x8 g1;
  g1[0] = (int)((1u << 16)      // data_size = 1 -> 2 bytes
              | (1u << 20)      // pad_enable
              | (3u << 22)      // pad_interval code 3 -> every 16 DWORDs (64B)
              | (3u << 25));    // pad_amount  code 3 -> 4 DWORDs (16B)
  g1[1] = (int)(32u << 16);     // tensor_dim0 = 32  (bits 79:48 -> [63:48])
  g1[2] = (int)(128u << 16);    // tensor_dim1 = 128 (bits 111:80 -> [95:80])
  g1[3] = (int)(32u << 16);     // tile_dim0 = 32    (bits 127:112)
  g1[4] = 128;                  // tile_dim1 = 128   (bits 143:128)
  g1[5] = 1024;                 // tensor_dim0_stride = K (bits 207:160, low 32)
  g1[6] = 0;
  g1[7] = 0;
  i32x4 z4 = {0, 0, 0, 0};                    // groups 2/3 unused (2D tensor)
  i32x8 z8 = {0, 0, 0, 0, 0, 0, 0, 0};        // extra group (6-arg toolchain form)
  __builtin_amdgcn_tensor_load_to_lds(g0, g1, z4, z4, z8, 0);
}
#endif

// ---------------- prep kernels ----------------
__global__ void cvt_bf16_kernel(const float* __restrict__ src,
                                unsigned short* __restrict__ dst, int n) {
  for (int i = blockIdx.x * blockDim.x + threadIdx.x; i < n;
       i += gridDim.x * blockDim.x)
    dst[i] = f2bfu(src[i]);
}

// W [K=1024][N=1024] f32  ->  Wt [N][K] bf16
__global__ void transpose_bf16_kernel(const float* __restrict__ W,
                                      unsigned short* __restrict__ Wt) {
  const int total = DMODEL * DMODEL;
  for (int i = blockIdx.x * blockDim.x + threadIdx.x; i < total;
       i += gridDim.x * blockDim.x) {
    int n = i >> 10;
    int k = i & 1023;
    Wt[i] = f2bfu(W[k * DMODEL + n]);
  }
}

// ---------------- tiled WMMA GEMM:  C = A(bf16)[M,1024] x Wt(bf16)[N,K]^T + bias ----
// TDM path: double-buffered LDS, DMA of tile k+1 overlaps WMMAs on tile k.
// out_mode 0: bf16 scatter [B,H,S,Dk]   (Q, K)
// out_mode 1: bf16 scatter [B,H,Dk,S]   (V transposed for PV)
// out_mode 2: f32 row-major [M,N]       (final output)
__global__ __launch_bounds__(256) void gemm_bf16_wmma_kernel(
    const unsigned short* __restrict__ Ap,
    const unsigned short* __restrict__ Btp,
    const float* __restrict__ bias,
    void* __restrict__ Cout, int out_mode)
{
  constexpr int K  = DMODEL;
  constexpr int LD = 40;                       // padded bf16 stride (80B): bank-safe, 16B aligned
  __shared__ __bf16 As[2][128][LD];
  __shared__ __bf16 Bs[2][128][LD];

  const __bf16* A  = (const __bf16*)Ap;
  const __bf16* Bt = (const __bf16*)Btp;

  const int tid  = threadIdx.x;
  const int lane = tid & 31;
  const int wave = tid >> 5;
  const int wm   = wave & 3;                   // m offset 32*wm
  const int wn   = wave >> 2;                  // n offset 64*wn
  const int lh   = lane >> 4;                  // lane half
  const int l16  = lane & 15;

  const int m0 = blockIdx.x * 128;
  const int n0 = blockIdx.y * 128;

#ifdef USE_TDM
  const unsigned ldsA[2] = {(unsigned)(size_t)(void*)&As[0][0][0],
                            (unsigned)(size_t)(void*)&As[1][0][0]};
  const unsigned ldsB[2] = {(unsigned)(size_t)(void*)&Bs[0][0][0],
                            (unsigned)(size_t)(void*)&Bs[1][0][0]};
#endif

  v8f acc[2][4];
#pragma unroll
  for (int mi = 0; mi < 2; ++mi)
#pragma unroll
    for (int ni = 0; ni < 4; ++ni) acc[mi][ni] = v8f_zero();

#ifdef USE_TDM
  // prologue: DMA tile 0 into buffer 0 (A by wave0, B by wave1)
  if (wave == 0)      tdm_load_tile(&A [(size_t)m0 * K], ldsA[0]);
  else if (wave == 1) tdm_load_tile(&Bt[(size_t)n0 * K], ldsB[0]);
#endif

  for (int k0 = 0; k0 < K; k0 += 32) {
    const int cur = (k0 >> 5) & 1;
#ifdef USE_TDM
    // issue DMA for tile k+1 into the other buffer, then wait only for tile k
    if (wave == 0) {
      if (k0 + 32 < K) {
        tdm_load_tile(&A[(size_t)m0 * K + k0 + 32], ldsA[cur ^ 1]);
        __builtin_amdgcn_s_wait_tensorcnt(1);
      } else {
        __builtin_amdgcn_s_wait_tensorcnt(0);
      }
    } else if (wave == 1) {
      if (k0 + 32 < K) {
        tdm_load_tile(&Bt[(size_t)n0 * K + k0 + 32], ldsB[cur ^ 1]);
        __builtin_amdgcn_s_wait_tensorcnt(1);
      } else {
        __builtin_amdgcn_s_wait_tensorcnt(0);
      }
    }
    __syncthreads();
#else
    // cooperative load: 128x32 bf16 tiles, 8-element (16B) chunks, 512 chunks each
#pragma unroll
    for (int it = 0; it < 2; ++it) {
      int c   = tid + it * 256;
      int row = c >> 2;
      int col = (c & 3) * 8;
      *(v8bf*)&As[0][row][col] = *(const v8bf*)&A [(size_t)(m0 + row) * K + k0 + col];
      *(v8bf*)&Bs[0][row][col] = *(const v8bf*)&Bt[(size_t)(n0 + row) * K + k0 + col];
      if (k0 + 32 < K) {                       // prefetch next k-tile (global_prefetch_b8)
        __builtin_prefetch(&A [(size_t)(m0 + row) * K + k0 + 32 + col], 0, 1);
        __builtin_prefetch(&Bt[(size_t)(n0 + row) * K + k0 + 32 + col], 0, 1);
      }
    }
    __syncthreads();
#endif

#ifdef USE_TDM
    const int buf = cur;
#else
    const int buf = 0;
#endif
    Frag a[2], b[4];
#pragma unroll
    for (int mi = 0; mi < 2; ++mi) {
      int row = wm * 32 + mi * 16 + l16;
      int c0  = lh ? 8 : 0;                    // A layout: lane<16 -> K{0..7,16..23}
      a[mi].h[0] = *(const v8bf*)&As[buf][row][c0];
      a[mi].h[1] = *(const v8bf*)&As[buf][row][c0 + 16];
    }
#pragma unroll
    for (int ni = 0; ni < 4; ++ni) {
      int row = wn * 64 + ni * 16 + l16;
      int c0  = lh ? 8 : 0;                    // B layout: lane<16 -> K{0..7,16..23}
      b[ni].h[0] = *(const v8bf*)&Bs[buf][row][c0];
      b[ni].h[1] = *(const v8bf*)&Bs[buf][row][c0 + 16];
    }
#pragma unroll
    for (int mi = 0; mi < 2; ++mi)
#pragma unroll
      for (int ni = 0; ni < 4; ++ni)
        acc[mi][ni] = __builtin_amdgcn_wmma_f32_16x16x32_bf16(
            false, a[mi].v, false, b[ni].v, (short)0, acc[mi][ni], false, false);
    __syncthreads();   // nobody still reads buf when its DMA overwrite is issued next iter
  }

  // epilogue: C layout -> VGPR r holds M = r + 8*lane_half, N = l16
#pragma unroll
  for (int mi = 0; mi < 2; ++mi)
#pragma unroll
    for (int ni = 0; ni < 4; ++ni)
#pragma unroll
      for (int r = 0; r < 8; ++r) {
        int gm = m0 + wm * 32 + mi * 16 + r + lh * 8;
        int gn = n0 + wn * 64 + ni * 16 + l16;
        float v = acc[mi][ni][r] + bias[gn];
        if (out_mode == 2) {
          ((float*)Cout)[(size_t)gm * DMODEL + gn] = v;
        } else {
          int bb = gm >> 11, s = gm & 2047;
          int h  = gn >> 6,  dk = gn & 63;
          size_t idx = (out_mode == 0)
              ? ((((size_t)bb * HEADS + h) * SEQ + s) * DK + dk)
              : ((((size_t)bb * HEADS + h) * DK + dk) * SEQ + s);
          ((unsigned short*)Cout)[idx] = f2bfu(v);
        }
      }
}

// ---------------- flash attention: per wave 16 queries, sweep keys in 64-tiles ----
__global__ __launch_bounds__(256) void attn_wmma_kernel(
    const unsigned short* __restrict__ Qp,   // [B,H,S,Dk] bf16
    const unsigned short* __restrict__ Kp,   // [B,H,S,Dk] bf16
    const unsigned short* __restrict__ Vp,   // [B,H,Dk,S] bf16
    unsigned short* __restrict__ Ob)         // [B*S, DMODEL] bf16
{
  __shared__ __bf16 plds[8][16][72];         // per-wave P staging (144B rows: 16B aligned)

  const float SCALE = 0.125f;                // 1/sqrt(64)
  const int tid  = threadIdx.x;
  const int lane = tid & 31;
  const int wave = tid >> 5;
  const int lh   = lane >> 4;
  const int l16  = lane & 15;

  const int bh = blockIdx.y;                 // b*16 + h
  const int b  = bh >> 4;
  const int h  = bh & 15;
  const int q0 = blockIdx.x * 128 + wave * 16;

  const __bf16* Qh = (const __bf16*)Qp + (size_t)bh * SEQ * DK;
  const __bf16* Kh = (const __bf16*)Kp + (size_t)bh * SEQ * DK;
  const __bf16* Vh = (const __bf16*)Vp + (size_t)bh * DK * SEQ;

  // Q A-fragments (K-dim = 64 -> 2 fragments)
  Frag qa[2];
#pragma unroll
  for (int kd = 0; kd < 2; ++kd) {
    int row = q0 + l16;
    int c0  = kd * 32 + (lh ? 8 : 0);
    qa[kd].h[0] = *(const v8bf*)&Qh[(size_t)row * DK + c0];
    qa[kd].h[1] = *(const v8bf*)&Qh[(size_t)row * DK + c0 + 16];
  }

  float m_run[8], l_run[8];
#pragma unroll
  for (int r = 0; r < 8; ++r) { m_run[r] = -1e30f; l_run[r] = 0.f; }
  v8f o[4];
#pragma unroll
  for (int nd = 0; nd < 4; ++nd) o[nd] = v8f_zero();

  for (int kk0 = 0; kk0 < SEQ; kk0 += 64) {
    // ---- scores S = Q K^T  (16 x 64 tile) ----
    v8f s[4];
#pragma unroll
    for (int ns = 0; ns < 4; ++ns) {
      s[ns] = v8f_zero();
#pragma unroll
      for (int kd = 0; kd < 2; ++kd) {
        Frag kb;
        int krow = kk0 + ns * 16 + l16;
        int c0   = kd * 32 + (lh ? 8 : 0);
        kb.h[0] = *(const v8bf*)&Kh[(size_t)krow * DK + c0];
        kb.h[1] = *(const v8bf*)&Kh[(size_t)krow * DK + c0 + 16];
        s[ns] = __builtin_amdgcn_wmma_f32_16x16x32_bf16(
            false, qa[kd].v, false, kb.v, (short)0, s[ns], false, false);
      }
    }
    // ---- online softmax (half-wave reductions match C layout) ----
#pragma unroll
    for (int r = 0; r < 8; ++r) {
      float mx = fmaxf(fmaxf(s[0][r], s[1][r]), fmaxf(s[2][r], s[3][r]));
#pragma unroll
      for (int off = 8; off > 0; off >>= 1) mx = fmaxf(mx, __shfl_xor(mx, off, 16));
      float mnew  = fmaxf(m_run[r], mx * SCALE);
      float alpha = __expf(m_run[r] - mnew);
      float psum  = 0.f;
      int rowl = r + lh * 8;
#pragma unroll
      for (int ns = 0; ns < 4; ++ns) {
        float p = __expf(s[ns][r] * SCALE - mnew);
        psum += p;
        plds[wave][rowl][ns * 16 + l16] = f2bf(p);
      }
#pragma unroll
      for (int off = 8; off > 0; off >>= 1) psum += __shfl_xor(psum, off, 16);
      l_run[r] = l_run[r] * alpha + psum;
      m_run[r] = mnew;
#pragma unroll
      for (int nd = 0; nd < 4; ++nd) o[nd][r] *= alpha;
    }
    // ---- reload P as A-fragments (same-wave LDS ops are in-order) ----
    Frag pa[2];
#pragma unroll
    for (int kd = 0; kd < 2; ++kd) {
      int c0 = kd * 32 + (lh ? 8 : 0);
      pa[kd].h[0] = *(const v8bf*)&plds[wave][l16][c0];
      pa[kd].h[1] = *(const v8bf*)&plds[wave][l16][c0 + 16];
    }
    // ---- O += P V  (V stored [Dk][S]: columns contiguous) ----
#pragma unroll
    for (int nd = 0; nd < 4; ++nd)
#pragma unroll
      for (int kd = 0; kd < 2; ++kd) {
        Frag vb;
        int dk = nd * 16 + l16;
        int c0 = kk0 + kd * 32 + (lh ? 8 : 0);
        vb.h[0] = *(const v8bf*)&Vh[(size_t)dk * SEQ + c0];
        vb.h[1] = *(const v8bf*)&Vh[(size_t)dk * SEQ + c0 + 16];
        o[nd] = __builtin_amdgcn_wmma_f32_16x16x32_bf16(
            false, pa[kd].v, false, vb.v, (short)0, o[nd], false, false);
      }
  }

  // ---- epilogue: O/l  -> Ob[B*S, DMODEL] bf16 ----
#pragma unroll
  for (int r = 0; r < 8; ++r) {
    float inv = 1.0f / l_run[r];
    int qrow  = q0 + r + lh * 8;
#pragma unroll
    for (int nd = 0; nd < 4; ++nd) {
      int col = h * DK + nd * 16 + l16;
      Ob[((size_t)(b * SEQ + qrow)) * DMODEL + col] = f2bfu(o[nd][r] * inv);
    }
  }
}

// ---------------- host launcher ----------------
extern "C" void kernel_launch(void* const* d_in, const int* in_sizes, int n_in,
                              void* d_out, int out_size, void* d_ws, size_t ws_size,
                              hipStream_t stream) {
  const float* x  = (const float*)d_in[0];
  const float* Wq = (const float*)d_in[1];
  const float* bq = (const float*)d_in[2];
  const float* Wk = (const float*)d_in[3];
  const float* bk = (const float*)d_in[4];
  const float* Wv = (const float*)d_in[5];
  const float* bv = (const float*)d_in[6];
  const float* Wo = (const float*)d_in[7];
  const float* bo = (const float*)d_in[8];

  char* ws = (char*)d_ws;
  unsigned short* xb  = (unsigned short*)(ws);                       // 16 MB
  unsigned short* Wtq = (unsigned short*)(ws + 16777216);            // 2 MB each
  unsigned short* Wtk = Wtq + 1048576;
  unsigned short* Wtv = Wtk + 1048576;
  unsigned short* Wto = Wtv + 1048576;
  unsigned short* Qb  = (unsigned short*)(ws + 25165824);            // 16 MB each
  unsigned short* Kb  = Qb + 8388608;
  unsigned short* Vt  = Kb + 8388608;
  unsigned short* Ob  = Vt + 8388608;

  cvt_bf16_kernel<<<2048, 256, 0, stream>>>(x, xb, MROWS * DMODEL);
  transpose_bf16_kernel<<<1024, 256, 0, stream>>>(Wq, Wtq);
  transpose_bf16_kernel<<<1024, 256, 0, stream>>>(Wk, Wtk);
  transpose_bf16_kernel<<<1024, 256, 0, stream>>>(Wv, Wtv);
  transpose_bf16_kernel<<<1024, 256, 0, stream>>>(Wo, Wto);

  dim3 gg(MROWS / 128, DMODEL / 128);
  gemm_bf16_wmma_kernel<<<gg, 256, 0, stream>>>(xb, Wtq, bq, Qb, 0);
  gemm_bf16_wmma_kernel<<<gg, 256, 0, stream>>>(xb, Wtk, bk, Kb, 0);
  gemm_bf16_wmma_kernel<<<gg, 256, 0, stream>>>(xb, Wtv, bv, Vt, 1);

  dim3 ga(SEQ / 128, BATCH * HEADS);
  attn_wmma_kernel<<<ga, 256, 0, stream>>>(Qb, Kb, Vt, Ob);

  gemm_bf16_wmma_kernel<<<gg, 256, 0, stream>>>(Ob, Wto, bo, d_out, 2);
}
